// OscillatorySSMLayer_30975304139614
// MI455X (gfx1250) — matrix operand order repaired and checked by reading, not compiled, not run on hardware
//
#include <hip/hip_runtime.h>

// ---------------------------------------------------------------------------
// Oscillatory SSM layer for MI455X (gfx1250, wave32, WMMA bf16).
// B=8, T=1024, D=1024, S=256.
// Pipeline:
//   1) pack_x:   x[B,T,D,2] f32 -> Xaug bf16 [8192 x 2048]  ([Xr | Xi])
//   2) pack_w1:  build Wbig bf16 [2048 x 2816] = [Bx_r | Bx_i | gate | Dx_r | Dx_i]
//   3) pack_wc:  build Wc  bf16 [512 x 2048]  = [[Cwr, Cwi], [-Cwi, Cwr]]
//   4) GEMM1 (WMMA bf16, async-LDS double-buffered): Out1 = Xaug @ Wbig
//   5) chunked associative scan over T (3 kernels) -> Haug bf16 [8192 x 512]
//   6) GEMM2 (WMMA bf16): Out2 = Haug @ Wc
//   7) epilogue: y = Ch + Dx; out = y_r*y + y  (interleaved [..,2] store)
// ---------------------------------------------------------------------------

typedef __bf16 bf16_t;
typedef __bf16 v8bf  __attribute__((ext_vector_type(8)));
typedef __bf16 v16bf __attribute__((ext_vector_type(16)));
typedef float  v8f   __attribute__((ext_vector_type(8)));

static constexpr int Bb = 8;
static constexpr int Tt = 1024;
static constexpr int Dd = 1024;
static constexpr int Ss = 256;
static constexpr int Mrows = Bb * Tt;          // 8192
static constexpr int K1 = 2 * Dd;              // 2048
static constexpr int N1 = 2 * Ss + Ss + 2*Dd;  // 2816 = [256 | 256 | 256 | 1024 | 1024]
static constexpr int K2 = 2 * Ss;              // 512
static constexpr int N2 = 2 * Dd;              // 2048
static constexpr int NC = 32;                  // scan chunks
static constexpr int TC = Tt / NC;             // 32 steps per chunk
static constexpr int OUT_ELEMS = Mrows * Dd * 2;   // 16777216

// Out1 column sections
static constexpr int C_BXR = 0;
static constexpr int C_BXI = Ss;
static constexpr int C_GATE = 2 * Ss;
static constexpr int C_DXR = 3 * Ss;
static constexpr int C_DXI = 3 * Ss + Dd;

__device__ __forceinline__ float sigf(float x) { return 1.0f / (1.0f + __expf(-x)); }

// ---------------------------------------------------------------------------
// bf16 WMMA GEMM: C[M,N] f32 = A[M,K] bf16 (row-major) @ B[K,N] bf16.
// 256 threads = 8 waves; block tile 128x128, K-step 32; wave tile 32x64.
// A tile staged via GLOBAL_LOAD_ASYNC_TO_LDS_B128 (ASYNCcnt), double-buffered
// LDS so the next K-tile streams in while WMMAs run on the current one.
// N, K are compile-time so store/frag addressing folds into immediates.
// ---------------------------------------------------------------------------
#define BMT 128
#define BNT 128
#define BKT 32
#define LDA_P 40   // padded LDS stride (elements); 80B rows -> conflict-free
#define LDB_P 40

template<int N, int K>
__global__ __launch_bounds__(256)
void gemm_bf16_wmma(const bf16_t* __restrict__ A, const bf16_t* __restrict__ Bm,
                    float* __restrict__ C)
{
    __shared__ bf16_t ldsA[2][BMT * LDA_P];   // [buf][row][k]
    __shared__ bf16_t ldsB[2][BNT * LDB_P];   // [buf][n][k] (transposed)

    const int tid    = threadIdx.x;
    const int lane   = tid & 31;
    const int wave   = tid >> 5;
    const int wr     = wave >> 1;          // 0..3 -> 32-row band
    const int wc     = wave & 1;           // 0..1 -> 64-col band
    const int laneLo = lane & 15;
    const int laneHi = lane >> 4;
    const int n0blk  = blockIdx.x * BNT;
    const int m0blk  = blockIdx.y * BMT;

    // per-thread staging coordinates (fixed across K-steps)
    const int rowA  = tid >> 1;            // 0..127
    const int halfA = (tid & 1) * 16;      // 0 / 16 (elements)
    const int kkB   = tid >> 3;            // 0..31
    const int ccB   = (tid & 7) * 16;      // 0..112

    const bf16_t* aptr = A + (size_t)(m0blk + rowA) * K + halfA;
    const bf16_t* bptr = Bm + (size_t)kkB * N + n0blk + ccB;

    auto stage = [&](int k0, int buf) {
        // ---- A tile: async global->LDS, 2 x b128 per thread (IOFFSET hits both sides)
        bf16_t* la = &ldsA[buf][rowA * LDA_P + halfA];
        unsigned laddr = (unsigned)(uintptr_t)la;          // low 32 bits = LDS offset
        const bf16_t* ga = aptr + k0;
        asm volatile("global_load_async_to_lds_b128 %0, %1, off"
                     :: "v"(laddr), "v"(ga) : "memory");
        asm volatile("global_load_async_to_lds_b128 %0, %1, off offset:16"
                     :: "v"(laddr), "v"(ga) : "memory");
        // ---- B tile: load 16 bf16 then transpose-scatter into LDS
        const v8bf* src = (const v8bf*)(bptr + (size_t)k0 * N);
        v8bf b0 = src[0];
        v8bf b1 = src[1];
#pragma unroll
        for (int j = 0; j < 8; ++j) {
            ldsB[buf][(ccB + j) * LDB_P + kkB]     = b0[j];
            ldsB[buf][(ccB + 8 + j) * LDB_P + kkB] = b1[j];
        }
    };

    v8f acc[2][4];
#pragma unroll
    for (int mi = 0; mi < 2; ++mi)
#pragma unroll
        for (int ni = 0; ni < 4; ++ni)
            acc[mi][ni] = {};

    stage(0, 0);
    asm volatile("s_wait_asynccnt 0x0" ::: "memory");
    __syncthreads();

    for (int k0 = 0; k0 < K; k0 += BKT) {
        const int buf = (k0 >> 5) & 1;
        if (k0 + BKT < K)
            stage(k0 + BKT, buf ^ 1);          // prefetch next tile into back buffer

        // ---- fragments ----
        // A 16x32 layout: lanes 0-15 (rows 0-15): K 0..7 & 16..23; lanes 16-31: K 8..15 & 24..31
        v16bf afrag[2];
#pragma unroll
        for (int mi = 0; mi < 2; ++mi) {
            const int r  = wr * 32 + mi * 16 + laneLo;
            const int ka = laneHi * 8;
            v8bf lo = *(const v8bf*)&ldsA[buf][r * LDA_P + ka];
            v8bf hi = *(const v8bf*)&ldsA[buf][r * LDA_P + 16 + ka];
            v16bf a;
#pragma unroll
            for (int j = 0; j < 8; ++j) { a[j] = lo[j]; a[8 + j] = hi[j]; }
            afrag[mi] = a;
        }
        // B 32x16 layout: lane holds column N=lane%16; lanes 0-15: K 0..7 & 16..23; 16-31: K 8..15 & 24..31
        v16bf bfrag[4];
#pragma unroll
        for (int ni = 0; ni < 4; ++ni) {
            const int n  = wc * 64 + ni * 16 + laneLo;
            const int kb = laneHi * 8;
            v8bf lo = *(const v8bf*)&ldsB[buf][n * LDB_P + kb];
            v8bf hi = *(const v8bf*)&ldsB[buf][n * LDB_P + kb + 16];
            v16bf b;
#pragma unroll
            for (int j = 0; j < 8; ++j) { b[j] = lo[j]; b[8 + j] = hi[j]; }
            bfrag[ni] = b;
        }

#pragma unroll
        for (int mi = 0; mi < 2; ++mi)
#pragma unroll
            for (int ni = 0; ni < 4; ++ni)
                acc[mi][ni] = __builtin_amdgcn_wmma_f32_16x16x32_bf16(
                    false, afrag[mi], false, bfrag[ni],
                    (short)0, acc[mi][ni], false, false);

        asm volatile("s_wait_asynccnt 0x0" ::: "memory");
        __syncthreads();
    }

    // ---- store: VGPR i -> row base+i (lanes<16) / base+8+i (lanes>=16), col = lane%16 ----
#pragma unroll
    for (int mi = 0; mi < 2; ++mi) {
        const int mbase = m0blk + wr * 32 + mi * 16 + laneHi * 8;
#pragma unroll
        for (int ni = 0; ni < 4; ++ni) {
            const int nidx = n0blk + wc * 64 + ni * 16 + laneLo;
            float* crow = C + (size_t)mbase * N + nidx;
#pragma unroll
            for (int i = 0; i < 8; ++i)
                crow[i * N] = acc[mi][ni][i];   // N is constexpr -> immediate offsets
        }
    }
}

// ---------------------------------------------------------------------------
// Packing kernels
// ---------------------------------------------------------------------------
__global__ __launch_bounds__(256)
void pack_x(const float* __restrict__ x, bf16_t* __restrict__ Xaug)
{
    const int idx = blockIdx.x * 256 + threadIdx.x;      // over 8192*1024
    const int n = idx >> 10, d = idx & 1023;
    const float2 v = ((const float2*)x)[(size_t)n * Dd + d];
    Xaug[(size_t)n * K1 + d]      = (bf16_t)v.x;
    Xaug[(size_t)n * K1 + Dd + d] = (bf16_t)v.y;
}

__global__ __launch_bounds__(256)
void pack_w1(const float* __restrict__ Bwr, const float* __restrict__ Bwi,
             const float* __restrict__ Dwr, const float* __restrict__ Dwi,
             const float* __restrict__ gW, bf16_t* __restrict__ Wbig)
{
    const int idx = blockIdx.x * 256 + threadIdx.x;      // over 2048*2816
    const int r = idx / N1, c = idx % N1;
    const bool lo = (r < Dd);
    const int rr = lo ? r : r - Dd;
    float v;
    if (c < C_BXI)           { const int s = c;          v = lo ?  Bwr[rr*Ss + s] : -Bwi[rr*Ss + s]; }
    else if (c < C_GATE)     { const int s = c - C_BXI;  v = lo ?  Bwi[rr*Ss + s] :  Bwr[rr*Ss + s]; }
    else if (c < C_DXR)      { const int s = c - C_GATE; v = gW[(size_t)r*Ss + s]; }
    else if (c < C_DXI)      { const int d = c - C_DXR;  v = lo ?  Dwr[rr*Dd + d] : -Dwi[rr*Dd + d]; }
    else                     { const int d = c - C_DXI;  v = lo ?  Dwi[rr*Dd + d] :  Dwr[rr*Dd + d]; }
    Wbig[(size_t)r * N1 + c] = (bf16_t)v;
}

__global__ __launch_bounds__(256)
void pack_wc(const float* __restrict__ Cwr, const float* __restrict__ Cwi,
             bf16_t* __restrict__ Wc)
{
    const int idx = blockIdx.x * 256 + threadIdx.x;      // over 512*2048
    const int r = idx >> 11, c = idx & 2047;
    const bool lo = (r < Ss);
    const int rr = lo ? r : r - Ss;
    float v;
    if (c < Dd) v = lo ?  Cwr[rr*Dd + c]        : -Cwi[rr*Dd + c];
    else        v = lo ?  Cwi[rr*Dd + (c - Dd)] :  Cwr[rr*Dd + (c - Dd)];
    Wc[(size_t)r * N2 + c] = (bf16_t)v;
}

// ---------------------------------------------------------------------------
// Chunked complex-affine scan:  h_t = alpha_t * h_{t-1} + beta_t
//   alpha = (1-g)*damp*e^{i theta},  beta = g*(bxr + i bxi)
// ---------------------------------------------------------------------------
__global__ __launch_bounds__(256)
void scan_chunk(const float* __restrict__ Out1,
                const float* __restrict__ theta, const float* __restrict__ damp_p,
                const float* __restrict__ gate_b,
                float* __restrict__ Ar, float* __restrict__ Ai,
                float* __restrict__ Cr, float* __restrict__ Ci)
{
    const int idx = blockIdx.x * 256 + threadIdx.x;      // B*NC*S = 65536
    const int s = idx % Ss;
    const int c = (idx / Ss) % NC;
    const int b = idx / (Ss * NC);
    const float dmp = 0.5f + 0.5f * sigf(damp_p[s]);
    const float ct = cosf(theta[s]) * dmp, st = sinf(theta[s]) * dmp;
    const float gb = gate_b[s];
    float aR = 1.0f, aI = 0.0f, cR = 0.0f, cI = 0.0f;
    for (int t = c * TC; t < (c + 1) * TC; ++t) {
        const size_t base = (size_t)(b * Tt + t) * N1;
        const float g   = sigf(Out1[base + C_GATE + s] + gb);
        const float bxr = Out1[base + C_BXR + s];
        const float bxi = Out1[base + C_BXI + s];
        const float ar = (1.0f - g) * ct, ai = (1.0f - g) * st;
        const float nAR = ar * aR - ai * aI, nAI = ar * aI + ai * aR;
        const float nCR = ar * cR - ai * cI + g * bxr;
        const float nCI = ar * cI + ai * cR + g * bxi;
        aR = nAR; aI = nAI; cR = nCR; cI = nCI;
    }
    const int o = (b * NC + c) * Ss + s;
    Ar[o] = aR; Ai[o] = aI; Cr[o] = cR; Ci[o] = cI;
}

__global__ __launch_bounds__(256)
void scan_carry(const float* __restrict__ h0,
                const float* __restrict__ Ar, const float* __restrict__ Ai,
                const float* __restrict__ Cr, const float* __restrict__ Ci,
                float* __restrict__ Hent, float* __restrict__ outTail)
{
    const int idx = blockIdx.x * 256 + threadIdx.x;      // B*S = 2048
    const int s = idx % Ss, b = idx / Ss;
    float hr = h0[(size_t)(b * Ss + s) * 2 + 0];
    float hi = h0[(size_t)(b * Ss + s) * 2 + 1];
    for (int c = 0; c < NC; ++c) {
        const int o = (b * NC + c) * Ss + s;
        Hent[(size_t)o * 2 + 0] = hr;
        Hent[(size_t)o * 2 + 1] = hi;
        const float aR = Ar[o], aI = Ai[o];
        const float nr = aR * hr - aI * hi + Cr[o];
        const float ni = aR * hi + aI * hr + Ci[o];
        hr = nr; hi = ni;
    }
    outTail[(size_t)(b * Ss + s) * 2 + 0] = hr;   // h_final [B,S,2]
    outTail[(size_t)(b * Ss + s) * 2 + 1] = hi;
}

__global__ __launch_bounds__(256)
void scan_expand(const float* __restrict__ Out1,
                 const float* __restrict__ theta, const float* __restrict__ damp_p,
                 const float* __restrict__ gate_b, const float* __restrict__ Hent,
                 bf16_t* __restrict__ Haug)
{
    const int idx = blockIdx.x * 256 + threadIdx.x;      // B*NC*S = 65536
    const int s = idx % Ss;
    const int c = (idx / Ss) % NC;
    const int b = idx / (Ss * NC);
    const float dmp = 0.5f + 0.5f * sigf(damp_p[s]);
    const float ct = cosf(theta[s]) * dmp, st = sinf(theta[s]) * dmp;
    const float gb = gate_b[s];
    const int o = (b * NC + c) * Ss + s;
    float hr = Hent[(size_t)o * 2 + 0];
    float hi = Hent[(size_t)o * 2 + 1];
    for (int t = c * TC; t < (c + 1) * TC; ++t) {
        const size_t base = (size_t)(b * Tt + t) * N1;
        const float g   = sigf(Out1[base + C_GATE + s] + gb);
        const float bxr = Out1[base + C_BXR + s];
        const float bxi = Out1[base + C_BXI + s];
        const float ar = (1.0f - g) * ct, ai = (1.0f - g) * st;
        const float nr = ar * hr - ai * hi + g * bxr;
        const float ni = ar * hi + ai * hr + g * bxi;
        hr = nr; hi = ni;
        const size_t hb = (size_t)(b * Tt + t) * K2;
        Haug[hb + s]      = (bf16_t)hr;
        Haug[hb + Ss + s] = (bf16_t)hi;
    }
}

// ---------------------------------------------------------------------------
// Epilogue: y = Ch + Dx; out = y_r*y + y  ([B,T,D,2] interleaved)
// ---------------------------------------------------------------------------
__global__ __launch_bounds__(256)
void epilogue(const float* __restrict__ Out2, const float* __restrict__ Out1,
              float* __restrict__ out)
{
    const int idx = blockIdx.x * 256 + threadIdx.x;      // 8192*1024
    const int n = idx >> 10, d = idx & 1023;
    const float chr = Out2[(size_t)n * N2 + d];
    const float chi = Out2[(size_t)n * N2 + Dd + d];
    const float dxr = Out1[(size_t)n * N1 + C_DXR + d];
    const float dxi = Out1[(size_t)n * N1 + C_DXI + d];
    const float yr = chr + dxr, yi = chi + dxi;
    float2 o;
    o.x = yr * yr + yr;
    o.y = yr * yi + yi;
    ((float2*)out)[(size_t)n * Dd + d] = o;
}

// ---------------------------------------------------------------------------
extern "C" void kernel_launch(void* const* d_in, const int* in_sizes, int n_in,
                              void* d_out, int out_size, void* d_ws, size_t ws_size,
                              hipStream_t stream)
{
    const float* x     = (const float*)d_in[0];
    const float* h0    = (const float*)d_in[1];
    const float* theta = (const float*)d_in[2];
    const float* dampp = (const float*)d_in[3];
    const float* Bwr   = (const float*)d_in[4];
    const float* Bwi   = (const float*)d_in[5];
    const float* Cwr   = (const float*)d_in[6];
    const float* Cwi   = (const float*)d_in[7];
    const float* Dwr   = (const float*)d_in[8];
    const float* Dwi   = (const float*)d_in[9];
    const float* gW    = (const float*)d_in[10];
    const float* gb    = (const float*)d_in[11];
    float* out = (float*)d_out;

    // Workspace carve-up (all section sizes are 256B multiples)
    char* w = (char*)d_ws;
    bf16_t* Xaug = (bf16_t*)w;  w += (size_t)Mrows * K1 * sizeof(bf16_t);   // 32 MB
    bf16_t* Wbig = (bf16_t*)w;  w += (size_t)K1 * N1 * sizeof(bf16_t);      // 11 MB
    float*  Out1 = (float*)w;   w += (size_t)Mrows * N1 * sizeof(float);    // 88 MB
    bf16_t* Haug = (bf16_t*)w;  w += (size_t)Mrows * K2 * sizeof(bf16_t);   // 8 MB
    bf16_t* Wc   = (bf16_t*)w;  w += (size_t)K2 * N2 * sizeof(bf16_t);      // 2 MB
    float*  Out2 = (float*)w;   w += (size_t)Mrows * N2 * sizeof(float);    // 64 MB
    float*  Ar   = (float*)w;   w += (size_t)Bb * NC * Ss * sizeof(float);
    float*  Ai   = (float*)w;   w += (size_t)Bb * NC * Ss * sizeof(float);
    float*  Cr   = (float*)w;   w += (size_t)Bb * NC * Ss * sizeof(float);
    float*  Ci   = (float*)w;   w += (size_t)Bb * NC * Ss * sizeof(float);
    float*  Hent = (float*)w;   w += (size_t)Bb * NC * Ss * 2 * sizeof(float);

    // 1-3: packing
    pack_x <<<(Mrows * Dd) / 256, 256, 0, stream>>>(x, Xaug);
    pack_w1<<<(K1 * N1) / 256, 256, 0, stream>>>(Bwr, Bwi, Dwr, Dwi, gW, Wbig);
    pack_wc<<<(K2 * N2) / 256, 256, 0, stream>>>(Cwr, Cwi, Wc);

    // 4: fused input GEMM -> [Bx_r | Bx_i | gate_pre | Dx_r | Dx_i]
    gemm_bf16_wmma<N1, K1><<<dim3(N1 / BNT, Mrows / BMT), 256, 0, stream>>>(
        Xaug, Wbig, Out1);

    // 5: chunked scan
    scan_chunk <<<(Bb * NC * Ss) / 256, 256, 0, stream>>>(
        Out1, theta, dampp, gb, Ar, Ai, Cr, Ci);
    scan_carry <<<(Bb * Ss) / 256, 256, 0, stream>>>(
        h0, Ar, Ai, Cr, Ci, Hent, out + OUT_ELEMS);
    scan_expand<<<(Bb * NC * Ss) / 256, 256, 0, stream>>>(
        Out1, theta, dampp, gb, Hent, Haug);

    // 6: output GEMM -> Ch
    gemm_bf16_wmma<N2, K2><<<dim3(N2 / BNT, Mrows / BMT), 256, 0, stream>>>(
        Haug, Wc, Out2);

    // 7: epilogue
    epilogue<<<(Mrows * Dd) / 256, 256, 0, stream>>>(Out2, Out1, out);
}
